// InternS1ProMoeDecoderLayer_83597243449697
// MI455X (gfx1250) — matrix operand order repaired and checked by reading, not compiled, not run on hardware
//
#include <hip/hip_runtime.h>

// ---------------- problem constants ----------------
#define TT      1024
#define HID     1024
#define NH      16
#define KVH     4
#define DH      64
#define NE      64
#define NG      8
#define TOPK    8
#define INTER   512
#define QKVW    ((NH + 2*KVH) * DH)   // 1536
#define NPAIR   (TT * TOPK)           // 8192 token-expert pairs (always exact)

#define LDS_PAD 40                    // 32 halves used + 8 pad -> 80B row stride (16B aligned,
                                      // bank stride 20, gcd(20,64)=4 -> 16 distinct banks/frag)

typedef __attribute__((ext_vector_type(16))) _Float16 v16h;
typedef __attribute__((ext_vector_type(8)))  _Float16 v8h;
typedef __attribute__((ext_vector_type(8)))  float    v8f;

// ---------------- rmsnorm over a 1024-wide row ----------------
__global__ void __launch_bounds__(256) rmsnorm_kernel(const float* __restrict__ x,
                                                      const float* __restrict__ w,
                                                      float* __restrict__ y, int ncols) {
    int row = blockIdx.x;
    int tid = threadIdx.x;
    const float* xr = x + (size_t)row * ncols;
    float ss = 0.f;
    for (int i = tid; i < ncols; i += 256) { float v = xr[i]; ss += v * v; }
    __shared__ float red[8];
    for (int off = 16; off > 0; off >>= 1) ss += __shfl_xor(ss, off);
    if ((tid & 31) == 0) red[tid >> 5] = ss;
    __syncthreads();
    if (tid < 8) {
        float v = red[tid];
        for (int off = 4; off > 0; off >>= 1) v += __shfl_xor(v, off);
        if (tid == 0) red[0] = v;
    }
    __syncthreads();
    float r = rsqrtf(red[0] / (float)ncols + 1e-6f);
    float* yr = y + (size_t)row * ncols;
    for (int i = tid; i < ncols; i += 256) yr[i] = xr[i] * r * w[i];
}

// ============ shared WMMA tile helpers (all GEMMs use the same 64x64x32 scheme) ============
// As[M][K] row-major f16 (padded); BsT[N][K] (B transposed at staging time, padded).
// A frag: two ds_load_b128 (K = half*8..+7, 16+half*8..+7).
// B frag: two ds_load_b128 (K = half*8..+7, 16+half*8..+7).

__device__ __forceinline__ v16h load_afrag(const _Float16* As, int mrow, int half) {
    const _Float16* base = As + (size_t)mrow * LDS_PAD;
    v8h lo = *(const v8h*)(base + half * 8);
    v8h hi = *(const v8h*)(base + 16 + half * 8);
    return __builtin_shufflevector(lo, hi, 0,1,2,3,4,5,6,7,8,9,10,11,12,13,14,15);
}

__device__ __forceinline__ v16h load_bfrag(const _Float16* BsT, int n, int half) {
    const _Float16* base = BsT + (size_t)n * LDS_PAD;
    v8h lo = *(const v8h*)(base + half * 8);
    v8h hi = *(const v8h*)(base + 16 + half * 8);
    return __builtin_shufflevector(lo, hi, 0,1,2,3,4,5,6,7,8,9,10,11,12,13,14,15);
}

__device__ __forceinline__ v8h cvt8(float4 a, float4 b) {
    v8h r;
    r[0]=(_Float16)a.x; r[1]=(_Float16)a.y; r[2]=(_Float16)a.z; r[3]=(_Float16)a.w;
    r[4]=(_Float16)b.x; r[5]=(_Float16)b.y; r[6]=(_Float16)b.z; r[7]=(_Float16)b.w;
    return r;
}

// ---------------- dense WMMA GEMM: C[M,N] = A[M,K] @ B[K,N] (+ addend) ----------------
__global__ void __launch_bounds__(256) gemm_kernel(const float* __restrict__ A,
                                                   const float* __restrict__ B,
                                                   float* __restrict__ C,
                                                   const float* __restrict__ addend,
                                                   int M, int N, int K) {
    __shared__ _Float16 As[64 * LDS_PAD];
    __shared__ _Float16 BsT[64 * LDS_PAD];
    int m0 = blockIdx.y * 64, n0 = blockIdx.x * 64;
    int tid = threadIdx.x, lane = tid & 31, wave = tid >> 5;
    int wr = wave >> 1, wc = wave & 1;
    int ar = tid >> 2, ac = (tid & 3) * 8;   // A: row 0..63, 8 cols
    int br = tid >> 3, bc = (tid & 7) * 8;   // B: row 0..31, 8 cols
    v8f c0 = {}, c1 = {};
    int half = lane >> 4;
    int l15  = lane & 15;
    int mrow = wr * 16 + l15;
    int nn   = wc * 32 + l15;
    for (int k0 = 0; k0 < K; k0 += 32) {
        const float4* ap = (const float4*)(A + (size_t)(m0 + ar) * K + k0 + ac);
        float4 av0 = ap[0], av1 = ap[1];
        const float4* bp = (const float4*)(B + (size_t)(k0 + br) * N + n0 + bc);
        float4 bv0 = bp[0], bv1 = bp[1];
        if (k0 + 32 < K) {
            __builtin_prefetch((const void*)(A + (size_t)(m0 + ar) * K + k0 + 32 + ac), 0, 1);
            __builtin_prefetch((const void*)(B + (size_t)(k0 + 32 + br) * N + n0 + bc), 0, 1);
        }
        *(v8h*)&As[(size_t)ar * LDS_PAD + ac] = cvt8(av0, av1);
        _Float16* bcol = &BsT[(size_t)bc * LDS_PAD + br];
        bcol[0*LDS_PAD]=(_Float16)bv0.x; bcol[1*LDS_PAD]=(_Float16)bv0.y;
        bcol[2*LDS_PAD]=(_Float16)bv0.z; bcol[3*LDS_PAD]=(_Float16)bv0.w;
        bcol[4*LDS_PAD]=(_Float16)bv1.x; bcol[5*LDS_PAD]=(_Float16)bv1.y;
        bcol[6*LDS_PAD]=(_Float16)bv1.z; bcol[7*LDS_PAD]=(_Float16)bv1.w;
        __syncthreads();
        v16h a  = load_afrag(As, mrow, half);
        v16h b0 = load_bfrag(BsT, nn, half);
        v16h b1 = load_bfrag(BsT, nn + 16, half);
        c0 = __builtin_amdgcn_wmma_f32_16x16x32_f16(false, a, false, b0, (short)0, c0, false, false);
        c1 = __builtin_amdgcn_wmma_f32_16x16x32_f16(false, a, false, b1, (short)0, c1, false, false);
        __syncthreads();
    }
    int col0 = n0 + wc * 32 + l15;
    #pragma unroll
    for (int v = 0; v < 8; ++v) {
        int row = m0 + wr * 16 + half * 8 + v;
        float x0 = c0[v], x1 = c1[v];
        if (addend) {
            x0 += addend[(size_t)row * N + col0];
            x1 += addend[(size_t)row * N + col0 + 16];
        }
        C[(size_t)row * N + col0]      = x0;
        C[(size_t)row * N + col0 + 16] = x1;
    }
}

// ---------------- per-head q/k rmsnorm + RoPE, in place on qkv ----------------
__global__ void __launch_bounds__(32) qkpost_kernel(float* __restrict__ qkv,
                                                    const int* __restrict__ pos,
                                                    const float* __restrict__ qw,
                                                    const float* __restrict__ kw) {
    int t = blockIdx.x, j = blockIdx.y, lane = threadIdx.x;
    float* row = qkv + (size_t)t * QKVW + j * DH;
    const float* w = (j < NH) ? qw : kw;
    float a = row[lane], b = row[lane + 32];
    float ss = a * a + b * b;
    for (int off = 16; off > 0; off >>= 1) ss += __shfl_xor(ss, off);
    float r = rsqrtf(ss * (1.f / 64.f) + 1e-6f);
    a = a * r * w[lane];
    b = b * r * w[lane + 32];
    float p = (float)pos[t];
    float f = p * __powf(10000.f, -(float)lane / 32.f);
    float c = __cosf(f), s = __sinf(f);
    row[lane]      = a * c - b * s;
    row[lane + 32] = b * c + a * s;
}

// ---------------- causal GQA attention, one wave per (t, head), online softmax ------
__global__ void __launch_bounds__(32) attn_kernel(const float* __restrict__ qkv,
                                                  float* __restrict__ attn) {
    int t = blockIdx.x, h = blockIdx.y, lane = threadIdx.x;
    int kv = h >> 2;                       // H/KVH = 4
    const float* qrow  = qkv + (size_t)t * QKVW + h * DH;
    const float* kbase = qkv + (size_t)(NH * DH) + kv * DH;
    const float* vbase = qkv + (size_t)((NH + KVH) * DH) + kv * DH;
    float q0 = qrow[lane], q1 = qrow[lane + 32];
    float m = -1e30f, ssum = 0.f, a0 = 0.f, a1 = 0.f;
    for (int s = 0; s <= t; ++s) {
        const float* kr = kbase + (size_t)s * QKVW;
        float part = q0 * kr[lane] + q1 * kr[lane + 32];
        for (int off = 16; off > 0; off >>= 1) part += __shfl_xor(part, off);
        float sc = part * 0.125f;          // 1/sqrt(64)
        float mn = fmaxf(m, sc);
        float corr = __expf(m - mn);
        float p = __expf(sc - mn);
        const float* vr = vbase + (size_t)s * QKVW;
        ssum = ssum * corr + p;
        a0 = a0 * corr + p * vr[lane];
        a1 = a1 * corr + p * vr[lane + 32];
        m = mn;
    }
    float inv = 1.f / ssum;
    attn[(size_t)t * (NH * DH) + h * DH + lane]      = a0 * inv;
    attn[(size_t)t * (NH * DH) + h * DH + lane + 32] = a1 * inv;
}

// ---------------- routing: softmax over 64, top-1 per group of 8 ----------------
__global__ void __launch_bounds__(256) route_kernel(const float* __restrict__ logits,
                                                    int* __restrict__ ids,
                                                    float* __restrict__ wts,
                                                    int* __restrict__ counts) {
    int t = blockIdx.x * blockDim.x + threadIdx.x;
    if (t >= TT) return;
    const float* L = logits + (size_t)t * NE;
    float mx = -1e30f;
    for (int e = 0; e < NE; ++e) mx = fmaxf(mx, L[e]);
    float pr[NE]; float s = 0.f;
    for (int e = 0; e < NE; ++e) { pr[e] = __expf(L[e] - mx); s += pr[e]; }
    float inv = 1.f / s;
    float tw[NG]; int ti[NG]; float wsum = 0.f;
    for (int g = 0; g < NG; ++g) {
        float best = -1.f; int bi = 0;
        for (int j = 0; j < 8; ++j) {
            float p = pr[g * 8 + j] * inv;
            if (p > best) { best = p; bi = j; }
        }
        tw[g] = best; ti[g] = g * 8 + bi; wsum += best;
    }
    float winv = 1.f / wsum;
    for (int g = 0; g < NG; ++g) {
        ids[t * NG + g] = ti[g];
        wts[t * NG + g] = tw[g] * winv;
        atomicAdd(&counts[ti[g]], 1);
    }
}

__global__ void zero_kernel(int* counts, int* cursor) {
    int i = threadIdx.x;
    if (i < NE) { counts[i] = 0; cursor[i] = 0; }
}

__global__ void scan_kernel(const int* __restrict__ counts, int* __restrict__ offs) {
    if (threadIdx.x == 0) {
        int a = 0;
        for (int e = 0; e < NE; ++e) { offs[e] = a; a += counts[e]; }
        offs[NE] = a;
    }
}

__global__ void __launch_bounds__(256) scatter_kernel(const int* __restrict__ ids,
                                                      const float* __restrict__ wts,
                                                      const int* __restrict__ offs,
                                                      int* __restrict__ cursor,
                                                      int* __restrict__ pair_token,
                                                      float* __restrict__ pair_w) {
    int i = blockIdx.x * blockDim.x + threadIdx.x;
    if (i >= NPAIR) return;
    int e = ids[i];
    int pos = offs[e] + atomicAdd(&cursor[e], 1);
    pair_token[pos] = i >> 3;
    pair_w[pos] = wts[i];
}

// ---------------- MoE GEMM1: gu[p, 0:1024] = h2[token(p)] @ Wgu[e]  (gathered A) ----
__global__ void __launch_bounds__(256) moe_gemm1_kernel(const float* __restrict__ h2,
                                                        const float* __restrict__ Wgu,
                                                        float* __restrict__ gu,
                                                        const int* __restrict__ offs,
                                                        const int* __restrict__ pair_token) {
    int e = blockIdx.y;
    int start = offs[e], end = offs[e + 1];
    int m0 = blockIdx.z * 64;
    if (m0 >= end - start) return;
    const int N = 2 * INTER, K = HID;
    int n0 = blockIdx.x * 64;
    const float* B = Wgu + (size_t)e * HID * (2 * INTER);
    __shared__ _Float16 As[64 * LDS_PAD];
    __shared__ _Float16 BsT[64 * LDS_PAD];
    int tid = threadIdx.x, lane = tid & 31, wave = tid >> 5;
    int wr = wave >> 1, wc = wave & 1;
    int ar = tid >> 2, ac = (tid & 3) * 8;
    int br = tid >> 3, bc = (tid & 7) * 8;
    int gm = start + m0 + ar;
    int token = (gm < end) ? pair_token[gm] : pair_token[start];
    v8f c0 = {}, c1 = {};
    int half = lane >> 4, l15 = lane & 15;
    int mrow = wr * 16 + l15;
    int nn   = wc * 32 + l15;
    for (int k0 = 0; k0 < K; k0 += 32) {
        const float4* ap = (const float4*)(h2 + (size_t)token * HID + k0 + ac);
        float4 av0 = ap[0], av1 = ap[1];
        const float4* bp = (const float4*)(B + (size_t)(k0 + br) * N + n0 + bc);
        float4 bv0 = bp[0], bv1 = bp[1];
        if (k0 + 32 < K) {
            __builtin_prefetch((const void*)(B + (size_t)(k0 + 32 + br) * N + n0 + bc), 0, 1);
        }
        *(v8h*)&As[(size_t)ar * LDS_PAD + ac] = cvt8(av0, av1);
        _Float16* bcol = &BsT[(size_t)bc * LDS_PAD + br];
        bcol[0*LDS_PAD]=(_Float16)bv0.x; bcol[1*LDS_PAD]=(_Float16)bv0.y;
        bcol[2*LDS_PAD]=(_Float16)bv0.z; bcol[3*LDS_PAD]=(_Float16)bv0.w;
        bcol[4*LDS_PAD]=(_Float16)bv1.x; bcol[5*LDS_PAD]=(_Float16)bv1.y;
        bcol[6*LDS_PAD]=(_Float16)bv1.z; bcol[7*LDS_PAD]=(_Float16)bv1.w;
        __syncthreads();
        v16h a  = load_afrag(As, mrow, half);
        v16h b0 = load_bfrag(BsT, nn, half);
        v16h b1 = load_bfrag(BsT, nn + 16, half);
        c0 = __builtin_amdgcn_wmma_f32_16x16x32_f16(false, a, false, b0, (short)0, c0, false, false);
        c1 = __builtin_amdgcn_wmma_f32_16x16x32_f16(false, a, false, b1, (short)0, c1, false, false);
        __syncthreads();
    }
    int col0 = n0 + wc * 32 + l15;
    #pragma unroll
    for (int v = 0; v < 8; ++v) {
        int prow = start + m0 + wr * 16 + half * 8 + v;
        if (prow < end) {
            gu[(size_t)prow * N + col0]      = c0[v];
            gu[(size_t)prow * N + col0 + 16] = c1[v];
        }
    }
}

// ---------------- silu(g) * u * pair_w -> act[p, 0:512] ----------------
__global__ void __launch_bounds__(256) act_kernel(const float* __restrict__ gu,
                                                  const float* __restrict__ pair_w,
                                                  float* __restrict__ act) {
    int i = blockIdx.x * blockDim.x + threadIdx.x;   // NPAIR * INTER threads
    int p = i >> 9, c = i & 511;
    float g = gu[(size_t)p * 1024 + c];
    float u = gu[(size_t)p * 1024 + 512 + c];
    float sg = g / (1.f + __expf(-g));
    act[i] = sg * u * pair_w[p];
}

// ---------------- MoE GEMM2: out[token(p)] += act[p] @ Wdn[e]  (scatter-add) --------
__global__ void __launch_bounds__(256) moe_gemm2_kernel(const float* __restrict__ act,
                                                        const float* __restrict__ Wdn,
                                                        float* __restrict__ out,
                                                        const int* __restrict__ offs,
                                                        const int* __restrict__ pair_token) {
    int e = blockIdx.y;
    int start = offs[e], end = offs[e + 1];
    int m0 = blockIdx.z * 64;
    if (m0 >= end - start) return;
    const int N = HID, K = INTER;
    int n0 = blockIdx.x * 64;
    const float* B = Wdn + (size_t)e * INTER * HID;
    __shared__ _Float16 As[64 * LDS_PAD];
    __shared__ _Float16 BsT[64 * LDS_PAD];
    int tid = threadIdx.x, lane = tid & 31, wave = tid >> 5;
    int wr = wave >> 1, wc = wave & 1;
    int ar = tid >> 2, ac = (tid & 3) * 8;
    int br = tid >> 3, bc = (tid & 7) * 8;
    int gm = start + m0 + ar;
    int arow = (gm < end) ? gm : start;              // clamp to avoid OOB reads
    v8f c0 = {}, c1 = {};
    int half = lane >> 4, l15 = lane & 15;
    int mrow = wr * 16 + l15;
    int nn   = wc * 32 + l15;
    for (int k0 = 0; k0 < K; k0 += 32) {
        const float4* ap = (const float4*)(act + (size_t)arow * K + k0 + ac);
        float4 av0 = ap[0], av1 = ap[1];
        const float4* bp = (const float4*)(B + (size_t)(k0 + br) * N + n0 + bc);
        float4 bv0 = bp[0], bv1 = bp[1];
        if (k0 + 32 < K) {
            __builtin_prefetch((const void*)(B + (size_t)(k0 + 32 + br) * N + n0 + bc), 0, 1);
        }
        *(v8h*)&As[(size_t)ar * LDS_PAD + ac] = cvt8(av0, av1);
        _Float16* bcol = &BsT[(size_t)bc * LDS_PAD + br];
        bcol[0*LDS_PAD]=(_Float16)bv0.x; bcol[1*LDS_PAD]=(_Float16)bv0.y;
        bcol[2*LDS_PAD]=(_Float16)bv0.z; bcol[3*LDS_PAD]=(_Float16)bv0.w;
        bcol[4*LDS_PAD]=(_Float16)bv1.x; bcol[5*LDS_PAD]=(_Float16)bv1.y;
        bcol[6*LDS_PAD]=(_Float16)bv1.z; bcol[7*LDS_PAD]=(_Float16)bv1.w;
        __syncthreads();
        v16h a  = load_afrag(As, mrow, half);
        v16h b0 = load_bfrag(BsT, nn, half);
        v16h b1 = load_bfrag(BsT, nn + 16, half);
        c0 = __builtin_amdgcn_wmma_f32_16x16x32_f16(false, a, false, b0, (short)0, c0, false, false);
        c1 = __builtin_amdgcn_wmma_f32_16x16x32_f16(false, a, false, b1, (short)0, c1, false, false);
        __syncthreads();
    }
    int col0 = n0 + wc * 32 + l15;
    #pragma unroll
    for (int v = 0; v < 8; ++v) {
        int prow = start + m0 + wr * 16 + half * 8 + v;
        if (prow < end) {
            int token = pair_token[prow];
            atomicAdd(&out[(size_t)token * HID + col0],      c0[v]);
            atomicAdd(&out[(size_t)token * HID + col0 + 16], c1[v]);
        }
    }
}

// ---------------- host-side orchestration ----------------
extern "C" void kernel_launch(void* const* d_in, const int* in_sizes, int n_in,
                              void* d_out, int out_size, void* d_ws, size_t ws_size,
                              hipStream_t stream) {
    (void)in_sizes; (void)n_in; (void)out_size; (void)ws_size;
    const int*   positions = (const int*)d_in[0];
    const float* hidden    = (const float*)d_in[1];
    const float* Wqkv      = (const float*)d_in[2];
    const float* Wo        = (const float*)d_in[3];
    const float* qw        = (const float*)d_in[4];
    const float* kw        = (const float*)d_in[5];
    const float* inw       = (const float*)d_in[6];
    const float* postw     = (const float*)d_in[7];
    const float* Wg        = (const float*)d_in[8];
    const float* Wgu       = (const float*)d_in[9];
    const float* Wdn       = (const float*)d_in[10];
    float* out = (float*)d_out;

    char* ws = (char*)d_ws;
    size_t off = 0;
    auto alloc = [&](size_t bytes) -> void* {
        void* p = ws + off;
        off = (off + bytes + 255) & ~(size_t)255;
        return p;
    };
    float* h          = (float*)alloc((size_t)TT * HID * 4);
    float* qkv        = (float*)alloc((size_t)TT * QKVW * 4);
    float* attn       = (float*)alloc((size_t)TT * NH * DH * 4);
    float* h2         = (float*)alloc((size_t)TT * HID * 4);
    float* logits     = (float*)alloc((size_t)TT * NE * 4);
    int*   ids        = (int*)  alloc((size_t)NPAIR * 4);
    float* wts        = (float*)alloc((size_t)NPAIR * 4);
    int*   counts     = (int*)  alloc(NE * 4);
    int*   offs       = (int*)  alloc((NE + 1) * 4);
    int*   cursor     = (int*)  alloc(NE * 4);
    int*   pair_token = (int*)  alloc((size_t)NPAIR * 4);
    float* pair_w     = (float*)alloc((size_t)NPAIR * 4);
    float* gu         = (float*)alloc((size_t)NPAIR * 2 * INTER * 4);
    float* actb       = (float*)alloc((size_t)NPAIR * INTER * 4);

    // 1) input rmsnorm
    rmsnorm_kernel<<<TT, 256, 0, stream>>>(hidden, inw, h, HID);
    // 2) qkv = h @ Wqkv   (1024 x 1536 x 1024)
    gemm_kernel<<<dim3(QKVW / 64, TT / 64), 256, 0, stream>>>(h, Wqkv, qkv, nullptr, TT, QKVW, HID);
    // 3) q/k rmsnorm + rope in place
    qkpost_kernel<<<dim3(TT, NH + KVH), 32, 0, stream>>>(qkv, positions, qw, kw);
    // 4) causal GQA attention
    attn_kernel<<<dim3(TT, NH), 32, 0, stream>>>(qkv, attn);
    // 5) out = hidden + attn @ Wo  (residual fused)
    gemm_kernel<<<dim3(HID / 64, TT / 64), 256, 0, stream>>>(attn, Wo, out, hidden, TT, HID, NH * DH);
    // 6) post rmsnorm
    rmsnorm_kernel<<<TT, 256, 0, stream>>>(out, postw, h2, HID);
    // 7) router logits = h2 @ Wg  (1024 x 64 x 1024)
    gemm_kernel<<<dim3(NE / 64, TT / 64), 256, 0, stream>>>(h2, Wg, logits, nullptr, TT, NE, HID);
    // 8) routing
    zero_kernel<<<1, 128, 0, stream>>>(counts, cursor);
    route_kernel<<<TT / 256, 256, 0, stream>>>(logits, ids, wts, counts);
    scan_kernel<<<1, 32, 0, stream>>>(counts, offs);
    scatter_kernel<<<NPAIR / 256, 256, 0, stream>>>(ids, wts, offs, cursor, pair_token, pair_w);
    // 9) expert-grouped MoE: gate_up -> silu*u*w -> down (scatter-add into out)
    moe_gemm1_kernel<<<dim3(2 * INTER / 64, NE, NPAIR / 64), 256, 0, stream>>>(h2, Wgu, gu, offs, pair_token);
    act_kernel<<<(NPAIR * INTER) / 256, 256, 0, stream>>>(gu, pair_w, actb);
    moe_gemm2_kernel<<<dim3(HID / 64, NE, NPAIR / 64), 256, 0, stream>>>(actb, Wdn, out, offs, pair_token);
}
